// SparseConvUNet_58188216926924
// MI455X (gfx1250) — matrix operand order repaired and checked by reading, not compiled, not run on hardware
//
#include <hip/hip_runtime.h>

// ---------------------------------------------------------------------------
// SparseConvUNet for MI455X (gfx1250): bf16 WMMA gathered implicit-GEMM conv,
// vectorized b128 fragment loads, 16x32 output tile per wave.
// ---------------------------------------------------------------------------

typedef __attribute__((ext_vector_type(16))) __bf16          v16bf;
typedef __attribute__((ext_vector_type(8)))  float           v8f;
typedef __attribute__((ext_vector_type(8)))  unsigned short  u16x8;
typedef __attribute__((ext_vector_type(16))) unsigned short  u16x16;

__device__ __forceinline__ float scu_bf2f(unsigned short b) {
    return __builtin_bit_cast(float, (unsigned int)b << 16);
}
__device__ __forceinline__ unsigned short scu_f2bf(float f) {
    unsigned int u = __builtin_bit_cast(unsigned int, f);
    u += 0x7FFFu + ((u >> 16) & 1u);          // round-to-nearest-even
    return (unsigned short)(u >> 16);
}

// ---------------------------------------------------------------------------
// Core: gathered sparse/dense 3x3x3 conv as implicit GEMM on WMMA bf16.
// One wave computes a 16(row) x 32(out-channel) tile; K = 27*CIN in chunks of
// 32. Per the 16-bit A fragment layout, a lane's 16 elements are two
// contiguous 8-K runs -> two global_load_b128 gathers. B is pre-transposed to
// [Cout, 27*CIN] so a lane's fragment is the same two 8-K runs -> two b128 loads.
// Output: raw f32 = acc / cnt[row] + bias[col].
// ---------------------------------------------------------------------------
template <int CIN>
__global__ __launch_bounds__(32)
void scunet_conv_wmma(const unsigned short* __restrict__ x,    // [Ncells, CIN] bf16
                      const unsigned short* __restrict__ wt,   // [Cout, 27*CIN] bf16 (transposed)
                      const float*          __restrict__ bias, // [Cout]
                      const int*            __restrict__ nbr,  // [27, Ncells]
                      const float*          __restrict__ cnt,  // [Ncells]
                      float*                __restrict__ raw,  // [Ncells, Cout]
                      int Ncells, int Cout)
{
    constexpr int KTOT = 27 * CIN;                 // multiple of 16
    const int lane = threadIdx.x & 31;
    const int m    = lane & 15;
    const int hi   = lane >> 4;                    // lane half (K-split)
    const int r    = blockIdx.x * 16 + m;          // A row (Ncells % 16 == 0)
    const int co0  = blockIdx.y * 32 + m;          // B columns (two tiles)
    const int co1  = co0 + 16;
    const bool ok0 = co0 < Cout;
    const bool ok1 = co1 < Cout;
    const unsigned short* wrow0 = wt + (size_t)(ok0 ? co0 : 0) * KTOT;
    const unsigned short* wrow1 = wt + (size_t)(ok1 ? co1 : 0) * KTOT;

    v8f acc0 = {}, acc1 = {};
    for (int kk = 0; kk < KTOT; kk += 32) {
        // ---- A fragment: two contiguous 8-K runs (never cross a tap) ----
        u16x8 alo = {}, ahi = {};
        {
            const int K0 = kk + hi * 8;
            if (K0 < KTOT) {
                const int t = K0 / CIN, c = K0 - t * CIN;
                const int idx = nbr[t * Ncells + r];
                if (idx >= 0)
                    alo = *(const u16x8*)(x + (size_t)idx * CIN + c);
            }
            const int K1 = kk + 16 + hi * 8;
            if (K1 < KTOT) {
                const int t = K1 / CIN, c = K1 - t * CIN;
                const int idx = nbr[t * Ncells + r];
                if (idx >= 0)
                    ahi = *(const u16x8*)(x + (size_t)idx * CIN + c);
            }
        }
        // ---- B fragments: two 8-K runs of a transposed-weight row (same K as A) ----
        u16x8 b0lo = {}, b0hi = {}, b1lo = {}, b1hi = {};
        {
            const int Kb = kk + hi * 8;
            if (Kb < KTOT) {
                if (ok0) b0lo = *(const u16x8*)(wrow0 + Kb);
                if (ok1) b1lo = *(const u16x8*)(wrow1 + Kb);
            }
            if (Kb + 16 < KTOT) {
                if (ok0) b0hi = *(const u16x8*)(wrow0 + Kb + 16);
                if (ok1) b1hi = *(const u16x8*)(wrow1 + Kb + 16);
            }
        }
        const u16x16 av  = __builtin_shufflevector(alo,  ahi,  0,1,2,3,4,5,6,7,8,9,10,11,12,13,14,15);
        const u16x16 bv0 = __builtin_shufflevector(b0lo, b0hi, 0,1,2,3,4,5,6,7,8,9,10,11,12,13,14,15);
        const u16x16 bv1 = __builtin_shufflevector(b1lo, b1hi, 0,1,2,3,4,5,6,7,8,9,10,11,12,13,14,15);
        const v16bf a  = __builtin_bit_cast(v16bf, av);
        acc0 = __builtin_amdgcn_wmma_f32_16x16x32_bf16(
            false, a, false, __builtin_bit_cast(v16bf, bv0), (short)0, acc0, false, false);
        acc1 = __builtin_amdgcn_wmma_f32_16x16x32_bf16(
            false, a, false, __builtin_bit_cast(v16bf, bv1), (short)0, acc1, false, false);
    }

    // ---- C/D layout: lane => col (lane&15), VGPR vr => row vr + 8*(lane>>4) ----
    if (ok0) {
        const float bv = bias[co0];
#pragma unroll
        for (int vr = 0; vr < 8; ++vr) {
            const int row = blockIdx.x * 16 + vr + 8 * hi;
            raw[(size_t)row * Cout + co0] = acc0[vr] / cnt[row] + bv;
        }
    }
    if (ok1) {
        const float bv = bias[co1];
#pragma unroll
        for (int vr = 0; vr < 8; ++vr) {
            const int row = blockIdx.x * 16 + vr + 8 * hi;
            raw[(size_t)row * Cout + co1] = acc1[vr] / cnt[row] + bv;
        }
    }
}

// ---------------------------------------------------------------------------
// Index-structure kernels.
// ---------------------------------------------------------------------------
__global__ void scunet_table_init(int* __restrict__ table, int n) {
    int i = blockIdx.x * blockDim.x + threadIdx.x;
    if (i < n) table[i] = -1;
}

__global__ void scunet_scatter_table(const int* __restrict__ xyz,
                                     const int* __restrict__ nvalid,
                                     int* __restrict__ table, int N, int D) {
    int n = blockIdx.x * blockDim.x + threadIdx.x;
    if (n >= N) return;
    if (n < *nvalid) {
        int x = xyz[n * 3 + 0], y = xyz[n * 3 + 1], z = xyz[n * 3 + 2];
        table[(x * D + y) * D + z] = n;
    }
}

__global__ void scunet_build_nbr_sparse(const int* __restrict__ xyz,
                                        const int* __restrict__ table,
                                        const int* __restrict__ nvalid,
                                        int* __restrict__ nbr,
                                        float* __restrict__ cnt,
                                        int* __restrict__ occ, int N, int D) {
    int n = blockIdx.x * blockDim.x + threadIdx.x;
    if (n >= N) return;
    int x = xyz[n * 3 + 0], y = xyz[n * 3 + 1], z = xyz[n * 3 + 2];
    int count = 0, k = 0;
    for (int dx = -1; dx <= 1; ++dx)
        for (int dy = -1; dy <= 1; ++dy)
            for (int dz = -1; dz <= 1; ++dz) {
                int nx = x + dx, ny = y + dy, nz = z + dz;
                int ri = -1;
                if (nx >= 0 && nx < D && ny >= 0 && ny < D && nz >= 0 && nz < D)
                    ri = table[(nx * D + ny) * D + nz];
                nbr[k * N + n] = ri;
                if (ri >= 0) ++count;
                ++k;
            }
    cnt[n] = count > 0 ? (float)count : 1.0f;
    occ[n] = (n < *nvalid) ? 1 : 0;
}

__global__ void scunet_build_nbr_dense(const int* __restrict__ occ,
                                       int* __restrict__ nbr,
                                       float* __restrict__ cnt, int Dd) {
    int Gd = Dd * Dd * Dd;
    int g = blockIdx.x * blockDim.x + threadIdx.x;
    if (g >= Gd) return;
    int z = g % Dd, y = (g / Dd) % Dd, x = g / (Dd * Dd);
    int count = 0, k = 0;
    for (int dx = -1; dx <= 1; ++dx)
        for (int dy = -1; dy <= 1; ++dy)
            for (int dz = -1; dz <= 1; ++dz) {
                int nx = x + dx, ny = y + dy, nz = z + dz;
                int ri = -1;
                if (nx >= 0 && nx < Dd && ny >= 0 && ny < Dd && nz >= 0 && nz < Dd) {
                    int ng = (nx * Dd + ny) * Dd + nz;
                    if (occ[ng]) ri = ng;
                }
                nbr[k * Gd + g] = ri;
                if (ri >= 0) ++count;
                ++k;
            }
    cnt[g] = count > 0 ? (float)count : 1.0f;
}

// ---------------------------------------------------------------------------
// Pooling (segment max), upsample+concat, voxel gather.
// ---------------------------------------------------------------------------
__global__ void scunet_pool_voxel(const unsigned short* __restrict__ e0, // [N,C]
                                  const int* __restrict__ table,        // level-0 hash
                                  unsigned short* __restrict__ f1,      // [G1,C]
                                  int* __restrict__ occ1,
                                  int D1, int C, int D0g) {
    int total = D1 * D1 * D1 * C;
    int i = blockIdx.x * blockDim.x + threadIdx.x;
    if (i >= total) return;
    int c = i % C, g = i / C;
    int z = g % D1, y = (g / D1) % D1, x = g / (D1 * D1);
    float best = -3.0e38f;
    int found = 0;
    for (int a = 0; a < 8; ++a) {
        int fx = 2 * x + (a >> 2), fy = 2 * y + ((a >> 1) & 1), fz = 2 * z + (a & 1);
        int idx = table[(fx * D0g + fy) * D0g + fz];
        if (idx >= 0) {
            found = 1;
            best = fmaxf(best, scu_bf2f(e0[(size_t)idx * C + c]));
        }
    }
    f1[(size_t)g * C + c] = found ? scu_f2bf(best) : (unsigned short)0;
    if (c == 0) occ1[g] = found;
}

__global__ void scunet_pool_dense(const unsigned short* __restrict__ xin, // fine [(2Dc)^3, C]
                                  const int* __restrict__ occin,
                                  unsigned short* __restrict__ y,         // coarse [Dc^3, C]
                                  int* __restrict__ occy, int Dc, int C) {
    int Df = 2 * Dc;
    int total = Dc * Dc * Dc * C;
    int i = blockIdx.x * blockDim.x + threadIdx.x;
    if (i >= total) return;
    int c = i % C, g = i / C;
    int z = g % Dc, yy = (g / Dc) % Dc, x = g / (Dc * Dc);
    float best = -3.0e38f;
    int found = 0;
    for (int a = 0; a < 8; ++a) {
        int fx = 2 * x + (a >> 2), fy = 2 * yy + ((a >> 1) & 1), fz = 2 * z + (a & 1);
        int fg = (fx * Df + fy) * Df + fz;
        if (occin[fg]) {
            found = 1;
            best = fmaxf(best, scu_bf2f(xin[(size_t)fg * C + c]));
        }
    }
    y[(size_t)g * C + c] = found ? scu_f2bf(best) : (unsigned short)0;
    if (c == 0) occy[g] = found;
}

__global__ void scunet_concat_up(const unsigned short* __restrict__ skip, int Cs,
                                 const unsigned short* __restrict__ coarse, int Cc,
                                 const int* __restrict__ occf,
                                 unsigned short* __restrict__ out, int Df) {
    int Ct = Cs + Cc;
    int total = Df * Df * Df * Ct;
    int i = blockIdx.x * blockDim.x + threadIdx.x;
    if (i >= total) return;
    int c = i % Ct, g = i / Ct;
    if (c < Cs) {
        out[(size_t)g * Ct + c] = skip[(size_t)g * Cs + c];
    } else {
        int z = g % Df, y = (g / Df) % Df, x = g / (Df * Df);
        int Dcrs = Df >> 1;
        int pg = ((x >> 1) * Dcrs + (y >> 1)) * Dcrs + (z >> 1);
        out[(size_t)g * Ct + c] =
            occf[g] ? coarse[(size_t)pg * Cc + (c - Cs)] : (unsigned short)0;
    }
}

__global__ void scunet_concat_voxel(const unsigned short* __restrict__ e0, int Ce,
                                    const unsigned short* __restrict__ d1, int Cd,
                                    const int* __restrict__ xyz,
                                    const int* __restrict__ nvalid,
                                    unsigned short* __restrict__ out, int N, int D1) {
    int Ct = Ce + Cd;
    int i = blockIdx.x * blockDim.x + threadIdx.x;
    if (i >= N * Ct) return;
    int c = i % Ct, n = i / Ct;
    if (c < Ce) {
        out[(size_t)n * Ct + c] = e0[(size_t)n * Ce + c];
    } else {
        unsigned short v = 0;
        if (n < *nvalid) {
            int px = xyz[n * 3 + 0] >> 1, py = xyz[n * 3 + 1] >> 1, pz = xyz[n * 3 + 2] >> 1;
            int pg = (px * D1 + py) * D1 + pz;
            v = d1[(size_t)pg * Cd + (c - Ce)];
        }
        out[(size_t)n * Ct + c] = v;
    }
}

// ---------------------------------------------------------------------------
// BN stats (one block per channel, LDS tree reduction) and postproc.
// ---------------------------------------------------------------------------
__global__ void scunet_bn_stats(const float* __restrict__ raw,
                                const int* __restrict__ nvalid,
                                float* __restrict__ stats, int N, int C) {
    __shared__ float sh[256], sh2[256];
    int c = blockIdx.x;
    int nv = *nvalid;
    if (nv > N) nv = N;
    float s = 0.f, s2 = 0.f;
    for (int n = threadIdx.x; n < nv; n += blockDim.x) {
        float v = raw[(size_t)n * C + c];
        s += v;
        s2 += v * v;
    }
    sh[threadIdx.x] = s;
    sh2[threadIdx.x] = s2;
    __syncthreads();
    for (int st = 128; st > 0; st >>= 1) {
        if ((int)threadIdx.x < st) {
            sh[threadIdx.x] += sh[threadIdx.x + st];
            sh2[threadIdx.x] += sh2[threadIdx.x + st];
        }
        __syncthreads();
    }
    if (threadIdx.x == 0) {
        stats[c] = sh[0];
        stats[C + c] = sh2[0];
    }
}

__global__ void scunet_postproc(const float* __restrict__ raw,
                                const int* __restrict__ occ,
                                const float* __restrict__ stats,
                                const float* __restrict__ gamma,
                                const float* __restrict__ beta,
                                const int* __restrict__ nvalid,
                                unsigned short* __restrict__ out_bf,
                                float* __restrict__ out_f32,
                                int Ncells, int C, int do_relu, int do_bn) {
    int i = blockIdx.x * blockDim.x + threadIdx.x;
    if (i >= Ncells * C) return;
    int c = i % C, n = i / C;
    float v = raw[i];
    if (do_bn) {
        float cntv = fmaxf((float)(*nvalid), 1.0f);
        float mu  = stats[c] / cntv;
        float var = stats[C + c] / cntv - mu * mu;
        v = (v - mu) * rsqrtf(var + 1e-3f) * gamma[c] + beta[c];
    }
    if (do_relu) v = fmaxf(v, 0.0f);
    if (!occ[n]) v = 0.0f;
    if (out_bf)  out_bf[i]  = scu_f2bf(v);
    if (out_f32) out_f32[i] = v;
}

__global__ void scunet_cast_f32_bf16(const float* __restrict__ in,
                                     unsigned short* __restrict__ out, int n) {
    int i = blockIdx.x * blockDim.x + threadIdx.x;
    if (i < n) out[i] = scu_f2bf(in[i]);
}

// Cast + transpose weights: w[K, Cout] (f32) -> wt[Cout, K] (bf16).
__global__ void scunet_wt_cast(const float* __restrict__ w,
                               unsigned short* __restrict__ wt,
                               int Ktot, int Cout) {
    int i = blockIdx.x * blockDim.x + threadIdx.x;
    if (i >= Ktot * Cout) return;
    int K = i / Cout, co = i - K * Cout;
    wt[(size_t)co * Ktot + K] = scu_f2bf(w[i]);
}

// ---------------------------------------------------------------------------
// Host side
// ---------------------------------------------------------------------------
static inline int scu_ceil(int a, int b) { return (a + b - 1) / b; }

static inline void scu_launch_conv(const unsigned short* x, const unsigned short* wt,
                                   const float* bias, const int* nbr, const float* cnt,
                                   float* raw, int Ncells, int Cin, int Cout,
                                   hipStream_t s) {
    dim3 grid((unsigned)scu_ceil(Ncells, 16), (unsigned)scu_ceil(Cout, 32), 1);
    dim3 blk(32, 1, 1);
    switch (Cin) {
#define SCU_CASE(CN) \
    case CN: scunet_conv_wmma<CN><<<grid, blk, 0, s>>>(x, wt, bias, nbr, cnt, raw, Ncells, Cout); break;
        SCU_CASE(16)  SCU_CASE(32)  SCU_CASE(64)  SCU_CASE(128)
        SCU_CASE(192) SCU_CASE(256) SCU_CASE(384) SCU_CASE(512)
#undef SCU_CASE
        default: break;
    }
}

extern "C" void kernel_launch(void* const* d_in, const int* in_sizes, int n_in,
                              void* d_out, int out_size, void* d_ws, size_t ws_size,
                              hipStream_t stream) {
    (void)in_sizes; (void)n_in; (void)out_size; (void)ws_size;
    const int B = 2, N = 65536, D0g = 64;
    const int G  = D0g * D0g * D0g;
    const int D1 = 32, G1 = D1 * D1 * D1;
    const int D2 = 16, G2 = D2 * D2 * D2;
    const int D3 = 8,  G3 = D3 * D3 * D3;
    const int TB = 256;

    const float* feats  = (const float*)d_in[0];
    const int*   xyz    = (const int*)d_in[1];
    const int*   nvalid = (const int*)d_in[2];

    // Weight tensors in setup_inputs() dict (insertion) order: {idx, Cin, Cout}.
    struct WSpec { int idx, cin, cout; };
    const WSpec W[16] = {
        {3, 16, 32},  {5, 32, 64},   {7, 64, 64},   {9, 64, 128},
        {11, 128, 128}, {13, 128, 256}, {15, 256, 256}, {17, 256, 256},
        {19, 512, 256}, {21, 256, 256}, {23, 384, 128}, {25, 128, 128},
        {27, 192, 64},  {29, 64, 64},   {31, 64, 64},   {35, 64, 21}};

    // ---- deterministic bump allocator over d_ws ----
    char* ws = (char*)d_ws;
    size_t off = 0;
    auto alloc = [&](size_t bytes) -> void* {
        void* p = ws + off;
        off += bytes;
        off = (off + 255) & ~(size_t)255;
        return p;
    };

    unsigned short* wbf[16];
    for (int i = 0; i < 16; ++i)
        wbf[i] = (unsigned short*)alloc((size_t)27 * W[i].cin * W[i].cout * 2);

    int*   table = (int*)alloc((size_t)(G + 1) * 4);
    int*   occ0  = (int*)alloc((size_t)N * 4);
    int*   nbr0  = (int*)alloc((size_t)27 * N * 4);
    float* cnt0  = (float*)alloc((size_t)N * 4);
    int*   occ1  = (int*)alloc((size_t)G1 * 4);
    int*   nbr1  = (int*)alloc((size_t)27 * G1 * 4);
    float* cnt1  = (float*)alloc((size_t)G1 * 4);
    int*   occ2  = (int*)alloc((size_t)G2 * 4);
    int*   nbr2  = (int*)alloc((size_t)27 * G2 * 4);
    float* cnt2  = (float*)alloc((size_t)G2 * 4);
    int*   occ3  = (int*)alloc((size_t)G3 * 4);
    int*   nbr3  = (int*)alloc((size_t)27 * G3 * 4);
    float* cnt3  = (float*)alloc((size_t)G3 * 4);

    unsigned short* x0   = (unsigned short*)alloc((size_t)N * 16 * 2);
    unsigned short* e0a  = (unsigned short*)alloc((size_t)N * 32 * 2);
    unsigned short* e0   = (unsigned short*)alloc((size_t)N * 64 * 2);
    unsigned short* f1   = (unsigned short*)alloc((size_t)G1 * 64 * 2);
    unsigned short* e1a  = (unsigned short*)alloc((size_t)G1 * 64 * 2);
    unsigned short* e1   = (unsigned short*)alloc((size_t)G1 * 128 * 2);
    unsigned short* f2   = (unsigned short*)alloc((size_t)G2 * 128 * 2);
    unsigned short* e2a  = (unsigned short*)alloc((size_t)G2 * 128 * 2);
    unsigned short* e2   = (unsigned short*)alloc((size_t)G2 * 256 * 2);
    unsigned short* f3   = (unsigned short*)alloc((size_t)G3 * 256 * 2);
    unsigned short* mida = (unsigned short*)alloc((size_t)G3 * 256 * 2);
    unsigned short* mid  = (unsigned short*)alloc((size_t)G3 * 256 * 2);
    unsigned short* cat2 = (unsigned short*)alloc((size_t)G2 * 512 * 2);
    unsigned short* d2a  = (unsigned short*)alloc((size_t)G2 * 256 * 2);
    unsigned short* d2   = (unsigned short*)alloc((size_t)G2 * 256 * 2);
    // cat1 (G1*384) and cat0 (N*192) have disjoint lifetimes -> share one region
    size_t catbytes = (size_t)G1 * 384 * 2;
    size_t cat0b = (size_t)N * 192 * 2;
    if (cat0b > catbytes) catbytes = cat0b;
    unsigned short* cat1 = (unsigned short*)alloc(catbytes);
    unsigned short* cat0 = cat1;
    unsigned short* d1a  = (unsigned short*)alloc((size_t)G1 * 128 * 2);
    unsigned short* d1   = (unsigned short*)alloc((size_t)G1 * 128 * 2);
    unsigned short* d0a  = (unsigned short*)alloc((size_t)N * 64 * 2);
    unsigned short* d0   = (unsigned short*)alloc((size_t)N * 64 * 2);
    unsigned short* h1   = (unsigned short*)alloc((size_t)N * 64 * 2);
    float* raw   = (float*)alloc((size_t)N * 64 * 4);   // max over all conv outputs
    float* stats = (float*)alloc((size_t)2 * 64 * 4);

    // ---- convert + transpose all weights to bf16 [Cout, 27*Cin] once ----
    for (int i = 0; i < 16; ++i) {
        int Ktot = 27 * W[i].cin;
        int nel = Ktot * W[i].cout;
        scunet_wt_cast<<<scu_ceil(nel, TB), TB, 0, stream>>>(
            (const float*)d_in[W[i].idx], wbf[i], Ktot, W[i].cout);
    }

    for (int sc = 0; sc < B; ++sc) {
        const float* featb = feats + (size_t)sc * N * 16;
        const int*   xyzb  = xyz + (size_t)sc * N * 3;
        const int*   nvb   = nvalid + sc;
        float*       outb  = (float*)d_out + (size_t)sc * N * 21;

        // level-0 hash + neighbor table
        scunet_table_init<<<scu_ceil(G + 1, TB), TB, 0, stream>>>(table, G + 1);
        scunet_scatter_table<<<scu_ceil(N, TB), TB, 0, stream>>>(xyzb, nvb, table, N, D0g);
        scunet_build_nbr_sparse<<<scu_ceil(N, TB), TB, 0, stream>>>(
            xyzb, table, nvb, nbr0, cnt0, occ0, N, D0g);
        scunet_cast_f32_bf16<<<scu_ceil(N * 16, TB), TB, 0, stream>>>(featb, x0, N * 16);

        // encoder_0 (sparse): 16->32->64
        scu_launch_conv(x0, wbf[0], (const float*)d_in[4], nbr0, cnt0, raw, N, 16, 32, stream);
        scunet_postproc<<<scu_ceil(N * 32, TB), TB, 0, stream>>>(
            raw, occ0, nullptr, nullptr, nullptr, nvb, e0a, nullptr, N, 32, 1, 0);
        scu_launch_conv(e0a, wbf[1], (const float*)d_in[6], nbr0, cnt0, raw, N, 32, 64, stream);
        scunet_postproc<<<scu_ceil(N * 64, TB), TB, 0, stream>>>(
            raw, occ0, nullptr, nullptr, nullptr, nvb, e0, nullptr, N, 64, 1, 0);

        // pool to level1 (dense 32^3) + neighbor table
        scunet_pool_voxel<<<scu_ceil(G1 * 64, TB), TB, 0, stream>>>(e0, table, f1, occ1, D1, 64, D0g);
        scunet_build_nbr_dense<<<scu_ceil(G1, TB), TB, 0, stream>>>(occ1, nbr1, cnt1, D1);

        // encoder_1: 64->64->128
        scu_launch_conv(f1, wbf[2], (const float*)d_in[8], nbr1, cnt1, raw, G1, 64, 64, stream);
        scunet_postproc<<<scu_ceil(G1 * 64, TB), TB, 0, stream>>>(
            raw, occ1, nullptr, nullptr, nullptr, nvb, e1a, nullptr, G1, 64, 1, 0);
        scu_launch_conv(e1a, wbf[3], (const float*)d_in[10], nbr1, cnt1, raw, G1, 64, 128, stream);
        scunet_postproc<<<scu_ceil(G1 * 128, TB), TB, 0, stream>>>(
            raw, occ1, nullptr, nullptr, nullptr, nvb, e1, nullptr, G1, 128, 1, 0);

        // pool to level2 (16^3)
        scunet_pool_dense<<<scu_ceil(G2 * 128, TB), TB, 0, stream>>>(e1, occ1, f2, occ2, D2, 128);
        scunet_build_nbr_dense<<<scu_ceil(G2, TB), TB, 0, stream>>>(occ2, nbr2, cnt2, D2);

        // encoder_2: 128->128->256
        scu_launch_conv(f2, wbf[4], (const float*)d_in[12], nbr2, cnt2, raw, G2, 128, 128, stream);
        scunet_postproc<<<scu_ceil(G2 * 128, TB), TB, 0, stream>>>(
            raw, occ2, nullptr, nullptr, nullptr, nvb, e2a, nullptr, G2, 128, 1, 0);
        scu_launch_conv(e2a, wbf[5], (const float*)d_in[14], nbr2, cnt2, raw, G2, 128, 256, stream);
        scunet_postproc<<<scu_ceil(G2 * 256, TB), TB, 0, stream>>>(
            raw, occ2, nullptr, nullptr, nullptr, nvb, e2, nullptr, G2, 256, 1, 0);

        // pool to level3 (8^3)
        scunet_pool_dense<<<scu_ceil(G3 * 256, TB), TB, 0, stream>>>(e2, occ2, f3, occ3, D3, 256);
        scunet_build_nbr_dense<<<scu_ceil(G3, TB), TB, 0, stream>>>(occ3, nbr3, cnt3, D3);

        // middle: 256->256->256
        scu_launch_conv(f3, wbf[6], (const float*)d_in[16], nbr3, cnt3, raw, G3, 256, 256, stream);
        scunet_postproc<<<scu_ceil(G3 * 256, TB), TB, 0, stream>>>(
            raw, occ3, nullptr, nullptr, nullptr, nvb, mida, nullptr, G3, 256, 1, 0);
        scu_launch_conv(mida, wbf[7], (const float*)d_in[18], nbr3, cnt3, raw, G3, 256, 256, stream);
        scunet_postproc<<<scu_ceil(G3 * 256, TB), TB, 0, stream>>>(
            raw, occ3, nullptr, nullptr, nullptr, nvb, mid, nullptr, G3, 256, 1, 0);

        // decoder_2: concat(e2, up(mid)) -> 512->256->256 @ 16^3
        scunet_concat_up<<<scu_ceil(G2 * 512, TB), TB, 0, stream>>>(e2, 256, mid, 256, occ2, cat2, D2);
        scu_launch_conv(cat2, wbf[8], (const float*)d_in[20], nbr2, cnt2, raw, G2, 512, 256, stream);
        scunet_postproc<<<scu_ceil(G2 * 256, TB), TB, 0, stream>>>(
            raw, occ2, nullptr, nullptr, nullptr, nvb, d2a, nullptr, G2, 256, 1, 0);
        scu_launch_conv(d2a, wbf[9], (const float*)d_in[22], nbr2, cnt2, raw, G2, 256, 256, stream);
        scunet_postproc<<<scu_ceil(G2 * 256, TB), TB, 0, stream>>>(
            raw, occ2, nullptr, nullptr, nullptr, nvb, d2, nullptr, G2, 256, 1, 0);

        // decoder_1: concat(e1, up(d2)) -> 384->128->128 @ 32^3
        scunet_concat_up<<<scu_ceil(G1 * 384, TB), TB, 0, stream>>>(e1, 128, d2, 256, occ1, cat1, D1);
        scu_launch_conv(cat1, wbf[10], (const float*)d_in[24], nbr1, cnt1, raw, G1, 384, 128, stream);
        scunet_postproc<<<scu_ceil(G1 * 128, TB), TB, 0, stream>>>(
            raw, occ1, nullptr, nullptr, nullptr, nvb, d1a, nullptr, G1, 128, 1, 0);
        scu_launch_conv(d1a, wbf[11], (const float*)d_in[26], nbr1, cnt1, raw, G1, 128, 128, stream);
        scunet_postproc<<<scu_ceil(G1 * 128, TB), TB, 0, stream>>>(
            raw, occ1, nullptr, nullptr, nullptr, nvb, d1, nullptr, G1, 128, 1, 0);

        // decoder_0 (sparse): concat(e0, gather(d1)) -> 192->64->64
        scunet_concat_voxel<<<scu_ceil(N * 192, TB), TB, 0, stream>>>(
            e0, 64, d1, 128, xyzb, nvb, cat0, N, D1);
        scu_launch_conv(cat0, wbf[12], (const float*)d_in[28], nbr0, cnt0, raw, N, 192, 64, stream);
        scunet_postproc<<<scu_ceil(N * 64, TB), TB, 0, stream>>>(
            raw, occ0, nullptr, nullptr, nullptr, nvb, d0a, nullptr, N, 64, 1, 0);
        scu_launch_conv(d0a, wbf[13], (const float*)d_in[30], nbr0, cnt0, raw, N, 64, 64, stream);
        scunet_postproc<<<scu_ceil(N * 64, TB), TB, 0, stream>>>(
            raw, occ0, nullptr, nullptr, nullptr, nvb, d0, nullptr, N, 64, 1, 0);

        // head1 (sparse, masked BN): 64->64
        scu_launch_conv(d0, wbf[14], (const float*)d_in[32], nbr0, cnt0, raw, N, 64, 64, stream);
        scunet_bn_stats<<<64, 256, 0, stream>>>(raw, nvb, stats, N, 64);
        scunet_postproc<<<scu_ceil(N * 64, TB), TB, 0, stream>>>(
            raw, occ0, stats, (const float*)d_in[33], (const float*)d_in[34], nvb,
            h1, nullptr, N, 64, 1, 1);

        // head2 (sparse, no relu): 64->21 -> f32 output
        scu_launch_conv(h1, wbf[15], (const float*)d_in[36], nbr0, cnt0, raw, N, 64, 21, stream);
        scunet_postproc<<<scu_ceil(N * 21, TB), TB, 0, stream>>>(
            raw, occ0, nullptr, nullptr, nullptr, nvb, nullptr, outb, N, 21, 0, 0);
    }
}
